// MultiHeadAttention_33990371180962
// MI455X (gfx1250) — matrix operand order repaired and checked by reading, not compiled, not run on hardware
//
#include <hip/hip_runtime.h>
#include <math.h>
#include <stdint.h>

// ---------------------------------------------------------------------------
// MI455X / gfx1250 multi-head attention forward.
//   B=4, S=2048, D=1024, H=16, DK=64
// Matrix math  : V_WMMA_F32_16X16X32_BF16 (bf16 in, f32 acc).
// GEMM staging : TENSOR_LOAD_TO_LDS (TDM, D# descriptor, s_wait_tensorcnt).
// Attn staging : GLOBAL_LOAD_ASYNC_TO_LDS_B128 (+ s_wait_asynccnt).
// ---------------------------------------------------------------------------

#define BATCH  4
#define SEQ    2048
#define DMODEL 1024
#define NHEAD  16
#define DHEAD  64

typedef __attribute__((ext_vector_type(16))) __bf16 v16bf;
typedef __attribute__((ext_vector_type(8)))  __bf16 v8bf;
typedef __attribute__((ext_vector_type(8)))  float  v8f;
typedef __attribute__((ext_vector_type(4)))  unsigned int v4u;
typedef __attribute__((ext_vector_type(8)))  int v8i;
typedef __attribute__((ext_vector_type(4)))  int v4i;

#define WMMA_BF16(A, B, C)                                                   \
  __builtin_amdgcn_wmma_f32_16x16x32_bf16(false, (A), false, (B), (short)0,  \
                                          (C), false, false)

// ---------------------------------------------------------------------------
// TDM: one-descriptor 2D tile load, global -> LDS, bf16 elements.
// Hardware pad: +16B after every 64B stored => LDS row stride 80B, which is
// exactly the [rows][40] padded layout the WMMA fragment loads expect.
// ---------------------------------------------------------------------------
static __device__ __forceinline__ void tdm_load_tile_2d(
    uint32_t lds_addr, const __bf16* gptr, uint32_t tile_rows,
    uint32_t tile_cols, uint32_t row_stride_elts, uint32_t tensor_rows) {
  uint64_t ga = (uint64_t)(uintptr_t)gptr;

  v4u g0;
  g0[0] = 1u;              // count = 1 valid descriptor, user mode
  g0[1] = lds_addr;        // LDS destination (bytes)
  g0[2] = (uint32_t)ga;    // global_addr[31:0]
  g0[3] = (uint32_t)((ga >> 32) & 0x1FFFFFFu) | (2u << 30);  // [56:32]|type=2

  const uint32_t td0 = row_stride_elts;  // tensor dim0 (elements, OOB only)
  v8i g1;
  g1[0] = (int)((1u << 16) |   // data_size = 1 -> 2-byte elements
                (1u << 20) |   // pad_enable (LDS destination padding)
                (3u << 22) |   // pad_interval = 16 DWORDs (64B)
                (3u << 25));   // pad_amount   =  4 DWORDs (16B)
  g1[1] = (int)((td0 & 0xFFFFu) << 16);                       // dim0[15:0]
  g1[2] = (int)((td0 >> 16) | ((tensor_rows & 0xFFFFu) << 16));
  g1[3] = (int)((tensor_rows >> 16) | (tile_cols << 16));     // tile_dim0
  g1[4] = (int)(tile_rows);                                   // tile_dim1
  g1[5] = (int)row_stride_elts;  // tensor_dim0_stride[31:0]
  g1[6] = 0;                     // stride[47:32] | dim1_stride[15:0]
  g1[7] = 0;

  v4i z4 = {0, 0, 0, 0};                     // groups 2/3 unused (2D tensor)
  v8i z8 = {0, 0, 0, 0, 0, 0, 0, 0};         // extra group (clang-23 arity)
  __builtin_amdgcn_tensor_load_to_lds(g0, g1, z4, z4, z8, 0);
}

// Two async 16B memory->LDS copies (32 contiguous bytes), GVS addressing:
//   LDS[lds(+16)] = MEM[saddr + goff(+16)]    (tracked by ASYNCcnt)
static __device__ __forceinline__ void async_cp32(uint32_t lds,
                                                  const __bf16* base,
                                                  uint32_t goff) {
  asm volatile(
      "global_load_async_to_lds_b128 %0, %2, %4\n\t"
      "global_load_async_to_lds_b128 %1, %3, %4"
      :
      : "v"(lds), "v"(lds + 16u), "v"(goff), "v"(goff + 16u), "s"(base)
      : "memory");
}

// Assemble a v16bf A/B fragment from two 16-byte chunks (ISA 7.12.2 layouts:
// 16-bit operands are two 8-element K-runs per lane).
static __device__ __forceinline__ v16bf frag_from(const __bf16* lo,
                                                  const __bf16* hi) {
  v8bf a = *(const v8bf*)lo;
  v8bf b = *(const v8bf*)hi;
  v16bf r;
#pragma unroll
  for (int i = 0; i < 8; ++i) { r[i] = a[i]; r[i + 8] = b[i]; }
  return r;
}

static __device__ __forceinline__ void store_out(float* p, float v) { *p = v; }
static __device__ __forceinline__ void store_out(__bf16* p, float v) {
  *p = (__bf16)v;
}

// ---------------------------------------------------------------------------
// Elementwise fp32 -> bf16 (one-shot; makes all GEMM operands raw-copyable).
// ---------------------------------------------------------------------------
__global__ __launch_bounds__(256) void f32_to_bf16_vec(
    const float* __restrict__ in, __bf16* __restrict__ out) {
  size_t i = ((size_t)blockIdx.x * 256 + threadIdx.x) * 8;
  float4 a = *(const float4*)(in + i);
  float4 b = *(const float4*)(in + i + 4);
  v8bf o;
  o[0] = (__bf16)a.x; o[1] = (__bf16)a.y; o[2] = (__bf16)a.z; o[3] = (__bf16)a.w;
  o[4] = (__bf16)b.x; o[5] = (__bf16)b.y; o[6] = (__bf16)b.z; o[7] = (__bf16)b.w;
  *(v8bf*)(out + i) = o;
}

// ---------------------------------------------------------------------------
// C[M,N] = A[M,K] @ W[N,K]^T + bias   (torch Linear convention, bf16 A/W)
// 128x128 block tile, BK=32, 8 waves x (32x64).
// Double-buffered staging: wave 0 issues two TDM descriptors per k-step.
// ---------------------------------------------------------------------------
template <typename OT>
__global__ __launch_bounds__(256) void gemm_xwT_bias(
    const __bf16* __restrict__ A, const __bf16* __restrict__ W,
    const float* __restrict__ bias, OT* __restrict__ Cout, int Mdim, int Ndim,
    int Kdim) {
  __shared__ __bf16 sA[2][128][40];  // 32 cols + 8 pad (80B rows == TDM pad)
  __shared__ __bf16 sB[2][128][40];  // rows are output columns (W rows)

  const int tid = threadIdx.x;
  const int lane = tid & 31;
  const int wid = tid >> 5;
  const int wm = wid & 3;   // 4 waves down M  (32 rows each)
  const int wn = wid >> 2;  // 2 waves across N (64 cols each)
  const int m0 = blockIdx.y * 128;
  const int n0 = blockIdx.x * 128;
  const int hf = lane >> 4;  // lane half (A/B layout split)
  const int l16 = lane & 15;

  const v8f vzero = {0.f, 0.f, 0.f, 0.f, 0.f, 0.f, 0.f, 0.f};
  v8f acc[2][4];
#pragma unroll
  for (int i = 0; i < 2; ++i)
#pragma unroll
    for (int j = 0; j < 4; ++j) acc[i][j] = vzero;

#define STAGE_TDM(buf, kk)                                                   \
  do {                                                                       \
    tdm_load_tile_2d((uint32_t)(size_t)&sA[buf][0][0],                       \
                     A + (size_t)m0 * Kdim + (kk), 128u, 32u,                \
                     (uint32_t)Kdim, (uint32_t)Mdim);                        \
    tdm_load_tile_2d((uint32_t)(size_t)&sB[buf][0][0],                       \
                     W + (size_t)n0 * Kdim + (kk), 128u, 32u,                \
                     (uint32_t)Kdim, (uint32_t)Ndim);                        \
  } while (0)

  const int nk = Kdim / 32;
  if (wid == 0) STAGE_TDM(0, 0);

  for (int kt = 0; kt < nk; ++kt) {
    const int buf = kt & 1;
    if (wid == 0) {
      if (kt + 1 < nk) {
        STAGE_TDM(buf ^ 1, (kt + 1) * 32);
        // two newest TDM ops (next tile) in flight; current tile retired
        __builtin_amdgcn_s_wait_tensorcnt(2);
      } else {
        __builtin_amdgcn_s_wait_tensorcnt(0);
      }
    }
    __syncthreads();  // publish tile `buf` to all 8 waves

    v16bf af[2], bfr[4];
#pragma unroll
    for (int i = 0; i < 2; ++i) {
      int r = wm * 32 + i * 16 + l16;
      int c0 = hf ? 8 : 0, c1 = hf ? 24 : 16;  // A 16x32 bf16 lane layout
      af[i] = frag_from(&sA[buf][r][c0], &sA[buf][r][c1]);
    }
#pragma unroll
    for (int j = 0; j < 4; ++j) {
      int c = wn * 64 + j * 16 + l16;
      int kb = hf ? 8 : 0;  // B 32x16 bf16 lane layout
      bfr[j] = frag_from(&sB[buf][c][kb], &sB[buf][c][kb + 16]);
    }
#pragma unroll
    for (int i = 0; i < 2; ++i)
#pragma unroll
      for (int j = 0; j < 4; ++j)
        acc[i][j] = WMMA_BF16(af[i], bfr[j], acc[i][j]);
    __syncthreads();  // everyone done reading buf before it is re-staged
  }
#undef STAGE_TDM

  // epilogue: C/D 16x16 f32 layout (col = l16, rows v + 8*half)
#pragma unroll
  for (int i = 0; i < 2; ++i)
#pragma unroll
    for (int j = 0; j < 4; ++j) {
      int col = n0 + wn * 64 + j * 16 + l16;
      float bv = bias[col];
#pragma unroll
      for (int v = 0; v < 8; ++v) {
        int row = m0 + wm * 32 + i * 16 + v + hf * 8;
        store_out(&Cout[(size_t)row * Ndim + col], acc[i][j][v] + bv);
      }
    }
}

// ---------------------------------------------------------------------------
// Flash attention, causal, per (b,h). Block = 128 query rows, 8 waves,
// each wave owns 16 query rows; online softmax over 64-key tiles.
// Q/K/V are projected bf16 in [B,S,D] layout (head h at cols h*64..h*64+63).
// K tiles staged with async-to-LDS; V staged manually (needs transpose).
// ---------------------------------------------------------------------------
__global__ __launch_bounds__(256) void mha_flash(const __bf16* __restrict__ Q,
                                                 const __bf16* __restrict__ K,
                                                 const __bf16* __restrict__ V,
                                                 __bf16* __restrict__ O) {
  __shared__ __bf16 sK[64][72];     // [key][dk]
  __shared__ __bf16 sVt[64][72];    // [dk][key]  (transposed on load)
  __shared__ __bf16 sP[8][16][72];  // per-wave P transpose patch

  const int tid = threadIdx.x;
  const int lane = tid & 31;
  const int wid = tid >> 5;
  const int hf = lane >> 4;
  const int l16 = lane & 15;

  const int qt = blockIdx.x;
  const int bh = blockIdx.y;
  const int b = bh / NHEAD;
  const int h = bh % NHEAD;
  const int qb0 = qt * 128;
  const int qrow0 = qb0 + wid * 16;

  const size_t base = ((size_t)b * SEQ) * DMODEL + (size_t)h * DHEAD;

  // Q fragments (16 rows x 64 dk = two 16x32 A-frags), scale 1/sqrt(64) folded
  v16bf qf[2];
  {
    const __bf16* qrow = Q + base + (size_t)(qrow0 + l16) * DMODEL;
#pragma unroll
    for (int f = 0; f < 2; ++f) {
      int c0 = f * 32 + (hf ? 8 : 0);
      int c1 = f * 32 + (hf ? 24 : 16);
      v8bf lo = *(const v8bf*)(qrow + c0);
      v8bf hi = *(const v8bf*)(qrow + c1);
#pragma unroll
      for (int i = 0; i < 8; ++i) {
        qf[f][i] = (__bf16)((float)lo[i] * 0.125f);
        qf[f][i + 8] = (__bf16)((float)hi[i] * 0.125f);
      }
    }
  }

  const v8f vzero = {0.f, 0.f, 0.f, 0.f, 0.f, 0.f, 0.f, 0.f};
  v8f m, l, accO[4];
#pragma unroll
  for (int v = 0; v < 8; ++v) { m[v] = -INFINITY; l[v] = 0.0f; }
#pragma unroll
  for (int j = 0; j < 4; ++j) accO[j] = vzero;

  const int nkb = qb0 / 64 + 2;  // causal: need keys 0 .. qb0+127

  const int ldr = tid >> 2;        // tile row 0..63
  const int ldc = (tid & 3) * 16;  // tile col segment

  for (int jb = 0; jb < nkb; ++jb) {
    {  // K tile: raw async copy (same layout).  V tile: manual transpose.
      async_cp32((uint32_t)(size_t)&sK[ldr][ldc], K,
                 (uint32_t)((base + (size_t)(jb * 64 + ldr) * DMODEL + ldc) *
                            2));
      const __bf16* vrow = V + base + (size_t)(jb * 64 + ldr) * DMODEL + ldc;
      v8bf v0 = *(const v8bf*)(vrow);
      v8bf v1 = *(const v8bf*)(vrow + 8);
#pragma unroll
      for (int i = 0; i < 8; ++i) {
        sVt[ldc + i][ldr] = v0[i];
        sVt[ldc + 8 + i][ldr] = v1[i];
      }
    }
    asm volatile("s_wait_asynccnt 0x0" ::: "memory");
    __syncthreads();

    // scores S = (Q*scale) x K^T : 4 tiles of 16 keys, 2 chained WMMAs each
    v8f s[4];
#pragma unroll
    for (int t = 0; t < 4; ++t) {
      int key = t * 16 + l16;
      int kb = hf ? 8 : 0;
      v16bf b0 = frag_from(&sK[key][kb], &sK[key][kb + 16]);
      v16bf b1 = frag_from(&sK[key][32 + kb], &sK[key][32 + kb + 16]);
      v8f z = vzero;
      z = WMMA_BF16(qf[0], b0, z);
      z = WMMA_BF16(qf[1], b1, z);
      s[t] = z;
    }

    // analytic causal mask (col > row -> -inf)
#pragma unroll
    for (int t = 0; t < 4; ++t) {
      int col = jb * 64 + t * 16 + l16;
#pragma unroll
      for (int v = 0; v < 8; ++v) {
        int row = qrow0 + v + hf * 8;
        if (col > row) s[t][v] = -INFINITY;
      }
    }

    // online softmax: xor-masks <=8 stay inside each 16-lane half,
    // matching the C-layout row split (rows v | v+8).
    v8f mnew, corr;
#pragma unroll
    for (int v = 0; v < 8; ++v) {
      float x = fmaxf(fmaxf(s[0][v], s[1][v]), fmaxf(s[2][v], s[3][v]));
      x = fmaxf(x, __shfl_xor(x, 1));
      x = fmaxf(x, __shfl_xor(x, 2));
      x = fmaxf(x, __shfl_xor(x, 4));
      x = fmaxf(x, __shfl_xor(x, 8));
      mnew[v] = fmaxf(m[v], x);
      corr[v] = __expf(m[v] - mnew[v]);  // first iter: exp(-inf) = 0
      m[v] = mnew[v];
    }
#pragma unroll
    for (int t = 0; t < 4; ++t)
#pragma unroll
      for (int v = 0; v < 8; ++v) s[t][v] = __expf(s[t][v] - mnew[v]);

#pragma unroll
    for (int v = 0; v < 8; ++v) {
      float x = s[0][v] + s[1][v] + s[2][v] + s[3][v];
      x += __shfl_xor(x, 1);
      x += __shfl_xor(x, 2);
      x += __shfl_xor(x, 4);
      x += __shfl_xor(x, 8);
      l[v] = l[v] * corr[v] + x;
    }
#pragma unroll
    for (int j = 0; j < 4; ++j)
#pragma unroll
      for (int v = 0; v < 8; ++v) accO[j][v] *= corr[v];

    // P: C-layout registers -> per-wave LDS patch -> A-layout fragments
#pragma unroll
    for (int t = 0; t < 4; ++t)
#pragma unroll
      for (int v = 0; v < 8; ++v)
        sP[wid][v + hf * 8][t * 16 + l16] = (__bf16)s[t][v];

    // same-wave LDS RAW: wait on the CDNA5 DS counter (no block barrier)
    asm volatile("s_wait_dscnt 0" ::: "memory");

    v16bf pf[2];
#pragma unroll
    for (int f = 0; f < 2; ++f) {
      int c0 = f * 32 + (hf ? 8 : 0);
      int c1 = f * 32 + (hf ? 24 : 16);
      pf[f] = frag_from(&sP[wid][l16][c0], &sP[wid][l16][c1]);
    }

    // O += P x V
#pragma unroll
    for (int j = 0; j < 4; ++j) {
      int dk = j * 16 + l16;
      int kb = hf ? 8 : 0;
      v16bf b0 = frag_from(&sVt[dk][kb], &sVt[dk][kb + 16]);
      v16bf b1 = frag_from(&sVt[dk][32 + kb], &sVt[dk][32 + kb + 16]);
      accO[j] = WMMA_BF16(pf[0], b0, accO[j]);
      accO[j] = WMMA_BF16(pf[1], b1, accO[j]);
    }
    __syncthreads();
  }

  // epilogue: O /= l, write bf16 concat layout [B,S,D]
#pragma unroll
  for (int j = 0; j < 4; ++j) {
    int col = j * 16 + l16;
#pragma unroll
    for (int v = 0; v < 8; ++v) {
      int row = qrow0 + v + hf * 8;
      float o = accO[j][v] / l[v];
      O[base + (size_t)row * DMODEL + col] = (__bf16)o;
    }
  }
}

// ---------------------------------------------------------------------------
extern "C" void kernel_launch(void* const* d_in, const int* in_sizes, int n_in,
                              void* d_out, int out_size, void* d_ws,
                              size_t ws_size, hipStream_t stream) {
  (void)in_sizes; (void)n_in; (void)out_size; (void)ws_size;

  const float* q = (const float*)d_in[0];
  const float* k = (const float*)d_in[1];
  const float* v = (const float*)d_in[2];
  // d_in[3] = causal mask: recomputed analytically inside mha_flash
  const float* Wq = (const float*)d_in[4];
  const float* bq = (const float*)d_in[5];
  const float* Wk = (const float*)d_in[6];
  const float* bk = (const float*)d_in[7];
  const float* Wv = (const float*)d_in[8];
  const float* bv = (const float*)d_in[9];
  const float* Wo = (const float*)d_in[10];
  const float* bo = (const float*)d_in[11];

  const int Mrows = BATCH * SEQ;  // 8192
  const size_t BSD = (size_t)Mrows * DMODEL;
  const size_t DD = (size_t)DMODEL * DMODEL;

  // workspace layout (bf16): converted inputs, projections, weights.
  // attention output reuses the converted-q slot (dead by then). ~104 MB.
  __bf16* qb = (__bf16*)d_ws;  // also `ao` later
  __bf16* kb = qb + BSD;
  __bf16* vb = kb + BSD;
  __bf16* qp = vb + BSD;
  __bf16* kp = qp + BSD;
  __bf16* vp = kp + BSD;
  __bf16* wq = vp + BSD;
  __bf16* wk = wq + DD;
  __bf16* wv = wk + DD;
  __bf16* wo = wv + DD;

  dim3 blk(256);

  // one-shot fp32 -> bf16 conversions (2048 elements per block)
  f32_to_bf16_vec<<<dim3(BSD / 2048), blk, 0, stream>>>(q, qb);
  f32_to_bf16_vec<<<dim3(BSD / 2048), blk, 0, stream>>>(k, kb);
  f32_to_bf16_vec<<<dim3(BSD / 2048), blk, 0, stream>>>(v, vb);
  f32_to_bf16_vec<<<dim3(DD / 2048), blk, 0, stream>>>(Wq, wq);
  f32_to_bf16_vec<<<dim3(DD / 2048), blk, 0, stream>>>(Wk, wk);
  f32_to_bf16_vec<<<dim3(DD / 2048), blk, 0, stream>>>(Wv, wv);
  f32_to_bf16_vec<<<dim3(DD / 2048), blk, 0, stream>>>(Wo, wo);

  dim3 gG(DMODEL / 128, Mrows / 128);  // (8, 64)
  gemm_xwT_bias<__bf16>
      <<<gG, blk, 0, stream>>>(qb, wq, bq, qp, Mrows, DMODEL, DMODEL);
  gemm_xwT_bias<__bf16>
      <<<gG, blk, 0, stream>>>(kb, wk, bk, kp, Mrows, DMODEL, DMODEL);
  gemm_xwT_bias<__bf16>
      <<<gG, blk, 0, stream>>>(vb, wv, bv, vp, Mrows, DMODEL, DMODEL);

  dim3 gA(SEQ / 128, BATCH * NHEAD);  // (16, 64)
  __bf16* ao = qb;                    // reuse converted-q slot
  mha_flash<<<gA, blk, 0, stream>>>(qp, kp, vp, ao);

  gemm_xwT_bias<float><<<gG, blk, 0, stream>>>(ao, wo, bo, (float*)d_out,
                                               Mrows, DMODEL, DMODEL);
}
